// LightGCL_20229295964574
// MI455X (gfx1250) — compile-verified, matrix operand order checked by reading, not run on hardware
//
#include <hip/hip_runtime.h>
#include <hip/hip_bf16.h>
#include <math.h>

#define NU_   200000
#define NI_   100000
#define DD    64
#define QQ    5
#define NEDGE 1000000
#define BB    1024
#define INV_TEMP 5.0f      // 1/0.2
#define LAM1_ 0.2f
#define LAM2_ 1e-07f
#define KEEP_P 0.9f        // 1 - DROP
#define EPS_  1e-12f

typedef __attribute__((ext_vector_type(16))) __bf16 v16bf;
typedef __attribute__((ext_vector_type(8)))  float  v8f;

// ---------------------------------------------------------------- utilities
__device__ __forceinline__ float hash_uniform(unsigned int a, unsigned int b) {
  unsigned long long x = (((unsigned long long)a) << 32) | (unsigned long long)b;
  x ^= x >> 33; x *= 0xff51afd7ed558ccdULL;
  x ^= x >> 33; x *= 0xc4ceb9fe1a85ec53ULL;
  x ^= x >> 33;
  return (float)(unsigned int)x * 2.3283064365386963e-10f; // [0,1)
}

__global__ void fill_zero_kernel(float* __restrict__ p, size_t n) {
  size_t i = (size_t)blockIdx.x * blockDim.x + threadIdx.x;
  size_t s = (size_t)gridDim.x * blockDim.x;
  for (; i < n; i += s) p[i] = 0.0f;
}

// ------------------------------------------------------- SpMM (edge gather/scatter)
// dst[row(e),:] += (vals[e]*mask(e)/keep) * src[col(e),:]   (wave per edge, float2/lane)
__global__ void spmm_kernel(const float* __restrict__ src,
                            float* __restrict__ dst,
                            const float* __restrict__ vals,
                            const int* __restrict__ dstIdx,
                            const int* __restrict__ srcIdx,
                            int nEdges, unsigned int seed) {
  int wpb  = blockDim.x >> 5;
  int gw   = blockIdx.x * wpb + (threadIdx.x >> 5);
  int lane = threadIdx.x & 31;
  int strd = gridDim.x * wpb;
  for (int e = gw; e < nEdges; e += strd) {
    float u = hash_uniform(seed, (unsigned int)e);
    if (u >= KEEP_P) continue;                 // wave-uniform
    float v = vals[e] * (1.0f / KEEP_P);
    int r = dstIdx[e], c = srcIdx[e];
    float2 x = reinterpret_cast<const float2*>(src + (size_t)c * DD)[lane];
    float* d = dst + (size_t)r * DD + lane * 2;
    atomicAdd(d,     v * x.x);
    atomicAdd(d + 1, v * x.y);
  }
}

// ------------------------------------------------------- T = wt(Q,N) @ X(N,64) -> (Q,64)
__global__ void lowrank_reduce_kernel(const float* __restrict__ wt,
                                      const float* __restrict__ X,
                                      float* __restrict__ T, int N) {
  int d = threadIdx.x & 63;
  int g = threadIdx.x >> 6;                    // 4 row groups per block
  float acc[QQ];
#pragma unroll
  for (int q = 0; q < QQ; ++q) acc[q] = 0.0f;
  for (int n = blockIdx.x * 4 + g; n < N; n += gridDim.x * 4) {
    float x = X[(size_t)n * DD + d];
#pragma unroll
    for (int q = 0; q < QQ; ++q) acc[q] += wt[(size_t)q * N + n] * x;
  }
  __shared__ float sh[4][QQ][DD];
#pragma unroll
  for (int q = 0; q < QQ; ++q) sh[g][q][d] = acc[q];
  __syncthreads();
  if (g == 0) {
#pragma unroll
    for (int q = 0; q < QQ; ++q)
      atomicAdd(&T[q * DD + d], sh[0][q][d] + sh[1][q][d] + sh[2][q][d] + sh[3][q][d]);
  }
}

// ------------------------------- G_sum += mul_s(N,Q)@T(Q,64) ; E_sum += Z (fused)
__global__ void apply_accum_kernel(const float* __restrict__ mul_s,
                                   const float* __restrict__ T,
                                   const float* __restrict__ Z,
                                   float* __restrict__ G_sum,
                                   float* __restrict__ E_sum, size_t N) {
  size_t total = N * DD;
  size_t strd  = (size_t)gridDim.x * blockDim.x;
  for (size_t idx = (size_t)blockIdx.x * blockDim.x + threadIdx.x; idx < total; idx += strd) {
    size_t n = idx >> 6; int d = (int)(idx & 63);
    float g = 0.0f;
#pragma unroll
    for (int q = 0; q < QQ; ++q) g += mul_s[n * QQ + q] * T[q * DD + d];
    G_sum[idx] += g;
    E_sum[idx] += Z[idx];
  }
}

// ------------------------------------------------------- row-normalize -> bf16 (wave/row)
__global__ void normalize_bf16_kernel(const float* __restrict__ X,
                                      __hip_bfloat16* __restrict__ Y, size_t N) {
  size_t row  = (size_t)blockIdx.x * (blockDim.x >> 5) + (threadIdx.x >> 5);
  int lane    = threadIdx.x & 31;
  size_t strd = (size_t)gridDim.x * (blockDim.x >> 5);
  for (; row < N; row += strd) {
    float2 x = reinterpret_cast<const float2*>(X + row * DD)[lane];
    float ss = x.x * x.x + x.y * x.y;
#pragma unroll
    for (int m = 1; m < 32; m <<= 1) ss += __shfl_xor(ss, m, 32);
    float inv = 1.0f / fmaxf(sqrtf(ss), EPS_);
    __hip_bfloat16* y = Y + row * DD + lane * 2;
    y[0] = __float2bfloat16(x.x * inv);
    y[1] = __float2bfloat16(x.y * inv);
  }
}

// -------------------- gather+normalize selected rows: G->bf16+f32, E->f32 (wave per b)
__global__ void gather_norm_kernel(const float* __restrict__ Gsum,
                                   const float* __restrict__ Esum,
                                   const int* __restrict__ ids,
                                   __hip_bfloat16* __restrict__ g_bf,
                                   float* __restrict__ g_f,
                                   float* __restrict__ e_f, int nB) {
  int b = blockIdx.x * (blockDim.x >> 5) + (threadIdx.x >> 5);
  int lane = threadIdx.x & 31;
  if (b >= nB) return;
  int row = ids[b];
  float2 gv = reinterpret_cast<const float2*>(Gsum + (size_t)row * DD)[lane];
  float ss = gv.x * gv.x + gv.y * gv.y;
#pragma unroll
  for (int m = 1; m < 32; m <<= 1) ss += __shfl_xor(ss, m, 32);
  float inv = 1.0f / fmaxf(sqrtf(ss), EPS_);
  float gx = gv.x * inv, gy = gv.y * inv;
  g_f[(size_t)b * DD + lane * 2]     = gx;
  g_f[(size_t)b * DD + lane * 2 + 1] = gy;
  g_bf[(size_t)b * DD + lane * 2]     = __float2bfloat16(gx);
  g_bf[(size_t)b * DD + lane * 2 + 1] = __float2bfloat16(gy);
  float2 ev = reinterpret_cast<const float2*>(Esum + (size_t)row * DD)[lane];
  float es = ev.x * ev.x + ev.y * ev.y;
#pragma unroll
  for (int m = 1; m < 32; m <<= 1) es += __shfl_xor(es, m, 32);
  float einv = 1.0f / fmaxf(sqrtf(es), EPS_);
  e_f[(size_t)b * DD + lane * 2]     = ev.x * einv;
  e_f[(size_t)b * DD + lane * 2 + 1] = ev.y * einv;
}

// ----------------------------------------- WMMA bf16 fragment load (A and B identical)
// per lane: 8 elems at kb + half*8  and  8 elems at kb + 16 + half*8   (row-major [*,64])
__device__ __forceinline__ v16bf load_frag(const __hip_bfloat16* __restrict__ base,
                                           int row, int kb, int half) {
  union { uint4 q[2]; v16bf v; } u;
  const __hip_bfloat16* r = base + (size_t)row * DD;
  u.q[0] = *reinterpret_cast<const uint4*>(r + kb + half * 8);
  u.q[1] = *reinterpret_cast<const uint4*>(r + kb + 16 + half * 8);
  return u.v;
}

// ----------- rowsum[b] += sum_n exp( dot(A[b,:],E[n,:]) / T )  via v_wmma_f32_16x16x32_bf16
// Software double-buffered: prefetch tile nt+stride (clamped index -> no divergence,
// EXEC stays all-1s) while the current tile's WMMA + exp execute.
__global__ void negscore_wmma_kernel(const __hip_bfloat16* __restrict__ A,     // [1024,64]
                                     const __hip_bfloat16* __restrict__ Enorm, // [N,64]
                                     float* __restrict__ rowsum, int nTiles) {
  int lane = threadIdx.x & 31;
  int wave = threadIdx.x >> 5;
  int half = lane >> 4;
  int m    = lane & 15;
  int rowTile = blockIdx.y;

  v16bf a0 = load_frag(A, rowTile * 16 + m, 0,  half);   // K = 0..31
  v16bf a1 = load_frag(A, rowTile * 16 + m, 32, half);   // K = 32..63

  float acc[8];
#pragma unroll
  for (int j = 0; j < 8; ++j) acc[j] = 0.0f;

  int wpb  = blockDim.x >> 5;
  int step = gridDim.x * wpb;
  int nt   = blockIdx.x * wpb + wave;

  if (nt < nTiles) {
    v16bf b0 = load_frag(Enorm, nt * 16 + m, 0,  half);
    v16bf b1 = load_frag(Enorm, nt * 16 + m, 32, half);
    while (true) {
      int nnt = nt + step;
      bool more = (nnt < nTiles);
      int  pnt  = more ? nnt : nt;                       // clamped prefetch index
      v16bf nb0 = load_frag(Enorm, pnt * 16 + m, 0,  half);
      v16bf nb1 = load_frag(Enorm, pnt * 16 + m, 32, half);

      v8f c = {0.f, 0.f, 0.f, 0.f, 0.f, 0.f, 0.f, 0.f};
      c = __builtin_amdgcn_wmma_f32_16x16x32_bf16(false, a0, false, b0, (short)0, c, false, false);
      c = __builtin_amdgcn_wmma_f32_16x16x32_bf16(false, a1, false, b1, (short)0, c, false, false);
#pragma unroll
      for (int j = 0; j < 8; ++j) acc[j] += __expf(c[j] * INV_TEMP);

      if (!more) break;                                  // wave-uniform
      b0 = nb0; b1 = nb1;
      nt = nnt;
    }
  }
  // reduce over N within each 16-lane half; VGPR j -> rows j (lanes 0-15), j+8 (lanes 16-31)
#pragma unroll
  for (int j = 0; j < 8; ++j) {
    float s = acc[j];
    s += __shfl_xor(s, 1, 32);
    s += __shfl_xor(s, 2, 32);
    s += __shfl_xor(s, 4, 32);
    s += __shfl_xor(s, 8, 32);
    if (m == 0) atomicAdd(&rowsum[rowTile * 16 + j + half * 8], s);
  }
}

// ------------------------------------------------------- small scalar kernels
__global__ void pos_score_kernel(const float* __restrict__ g, const float* __restrict__ e,
                                 float* __restrict__ acc) {
  int b = blockIdx.x * (blockDim.x >> 5) + (threadIdx.x >> 5);
  int lane = threadIdx.x & 31;
  if (b >= BB) return;
  float2 gv = reinterpret_cast<const float2*>(g + (size_t)b * DD)[lane];
  float2 ev = reinterpret_cast<const float2*>(e + (size_t)b * DD)[lane];
  float s = gv.x * ev.x + gv.y * ev.y;
#pragma unroll
  for (int m = 1; m < 32; m <<= 1) s += __shfl_xor(s, m, 32);
  if (lane == 0) atomicAdd(acc, fminf(fmaxf(s * INV_TEMP, -5.0f), 5.0f));
}

__global__ void bpr_kernel(const float* __restrict__ Eu, const float* __restrict__ Ei,
                           const int* __restrict__ uids, const int* __restrict__ pos,
                           const int* __restrict__ neg, float* __restrict__ acc) {
  int b = blockIdx.x * (blockDim.x >> 5) + (threadIdx.x >> 5);
  int lane = threadIdx.x & 31;
  if (b >= BB) return;
  int u = uids[b], p = pos[b], n = neg[b];
  float2 uv = reinterpret_cast<const float2*>(Eu + (size_t)u * DD)[lane];
  float2 pv = reinterpret_cast<const float2*>(Ei + (size_t)p * DD)[lane];
  float2 nv = reinterpret_cast<const float2*>(Ei + (size_t)n * DD)[lane];
  float sp = uv.x * pv.x + uv.y * pv.y;
  float sn = uv.x * nv.x + uv.y * nv.y;
#pragma unroll
  for (int m = 1; m < 32; m <<= 1) { sp += __shfl_xor(sp, m, 32); sn += __shfl_xor(sn, m, 32); }
  if (lane == 0) {
    float d = sp - sn;                                     // softplus(-d) = -log_sigmoid(d)
    atomicAdd(acc, fmaxf(-d, 0.0f) + log1pf(__expf(-fabsf(d))));
  }
}

__global__ void sumsq_kernel(const float* __restrict__ x, size_t n, float* __restrict__ acc) {
  __shared__ float sh[256];
  float s = 0.0f;
  for (size_t i = (size_t)blockIdx.x * blockDim.x + threadIdx.x; i < n;
       i += (size_t)gridDim.x * blockDim.x) { float v = x[i]; s += v * v; }
  sh[threadIdx.x] = s; __syncthreads();
  for (int off = 128; off > 0; off >>= 1) {
    if ((int)threadIdx.x < off) sh[threadIdx.x] += sh[threadIdx.x + off];
    __syncthreads();
  }
  if (threadIdx.x == 0) atomicAdd(acc, sh[0]);
}

__global__ void finalize_kernel(const float* __restrict__ rsU, const float* __restrict__ rsI,
                                const float* __restrict__ S, float* __restrict__ out) {
  __shared__ float shu[256], shi[256];
  float lu = 0.0f, li = 0.0f;
  for (int b = threadIdx.x; b < BB; b += 256) {
    lu += logf(rsU[b] + 1e-8f);
    li += logf(rsI[b] + 1e-8f);
  }
  shu[threadIdx.x] = lu; shi[threadIdx.x] = li; __syncthreads();
  for (int off = 128; off > 0; off >>= 1) {
    if ((int)threadIdx.x < off) { shu[threadIdx.x] += shu[threadIdx.x + off];
                                  shi[threadIdx.x] += shi[threadIdx.x + off]; }
    __syncthreads();
  }
  if (threadIdx.x == 0) {
    const float invB = 1.0f / (float)BB;
    float neg_score = shu[0] * invB + shi[0] * invB;
    float pos_score = (S[0] + S[1]) * invB;
    float loss_s   = -pos_score + neg_score;
    float loss_r   = S[2] * invB;
    float loss_reg = LAM2_ * S[3];
    float lam_s    = LAM1_ * loss_s;
    out[0] = loss_r + loss_reg + lam_s;
    out[1] = loss_r;
    out[2] = lam_s;
  }
}

// ---------------------------------------------------------------- launcher
extern "C" void kernel_launch(void* const* d_in, const int* in_sizes, int n_in,
                              void* d_out, int out_size, void* d_ws, size_t ws_size,
                              hipStream_t stream) {
  (void)in_sizes; (void)n_in; (void)out_size; (void)ws_size;
  const float* E_u_0   = (const float*)d_in[0];
  const float* E_i_0   = (const float*)d_in[1];
  const float* u_mul_s = (const float*)d_in[2];
  const float* v_mul_s = (const float*)d_in[3];
  const float* ut      = (const float*)d_in[4];
  const float* vt      = (const float*)d_in[5];
  const float* adj_v   = (const float*)d_in[6];
  const int*   adj_r   = (const int*)d_in[7];
  const int*   adj_c   = (const int*)d_in[8];
  const int*   uids    = (const int*)d_in[9];
  const int*   iids    = (const int*)d_in[10];
  const int*   pos     = (const int*)d_in[11];
  const int*   neg     = (const int*)d_in[12];
  float* out = (float*)d_out;

  const size_t EU = (size_t)NU_ * DD, EI = (size_t)NI_ * DD;
  float* ws      = (float*)d_ws;
  float* EU_SUM  = ws;
  float* EI_SUM  = EU_SUM + EU;
  float* GU_SUM  = EI_SUM + EI;
  float* GI_SUM  = GU_SUM + EU;
  float* ZU_A    = GI_SUM + EI;
  float* ZI_A    = ZU_A + EU;
  float* ZU_B    = ZI_A + EI;
  float* ZI_B    = ZU_B + EU;
  float* T_I     = ZI_B + EI;          // 320 (padded 512)
  float* T_U     = T_I + 512;          // 320 (padded 512)
  float* GSELF_U = T_U + 512;          // 1024*64
  float* ESELF_U = GSELF_U + BB * DD;
  float* GSELF_I = ESELF_U + BB * DD;
  float* ESELF_I = GSELF_I + BB * DD;
  float* RS_U    = ESELF_I + BB * DD;  // 1024
  float* RS_I    = RS_U + BB;          // 1024
  float* SCAL    = RS_I + BB;          // 16: [0]=posU [1]=posI [2]=bpr [3]=reg
  __hip_bfloat16* EU_NORM = (__hip_bfloat16*)(SCAL + 16);
  __hip_bfloat16* EI_NORM = EU_NORM + EU;
  __hip_bfloat16* GSELB_U = EI_NORM + EI;
  __hip_bfloat16* GSELB_I = GSELB_U + BB * DD;

  // seed accumulator sums with layer-0 embeddings
  hipMemcpyAsync(EU_SUM, E_u_0, EU * sizeof(float), hipMemcpyDeviceToDevice, stream);
  hipMemcpyAsync(GU_SUM, E_u_0, EU * sizeof(float), hipMemcpyDeviceToDevice, stream);
  hipMemcpyAsync(EI_SUM, E_i_0, EI * sizeof(float), hipMemcpyDeviceToDevice, stream);
  hipMemcpyAsync(GI_SUM, E_i_0, EI * sizeof(float), hipMemcpyDeviceToDevice, stream);

  // ---- layer 1 (prev = inputs) -> ZU_A/ZI_A
  fill_zero_kernel<<<2048, 256, 0, stream>>>(ZU_A, EU + EI);
  fill_zero_kernel<<<4, 256, 0, stream>>>(T_I, 1024);
  spmm_kernel<<<2048, 256, 0, stream>>>(E_i_0, ZU_A, adj_v, adj_r, adj_c, NEDGE, 2u);
  spmm_kernel<<<2048, 256, 0, stream>>>(E_u_0, ZI_A, adj_v, adj_c, adj_r, NEDGE, 3u);
  lowrank_reduce_kernel<<<512, 256, 0, stream>>>(vt, E_i_0, T_I, NI_);
  lowrank_reduce_kernel<<<512, 256, 0, stream>>>(ut, E_u_0, T_U, NU_);
  apply_accum_kernel<<<2048, 256, 0, stream>>>(u_mul_s, T_I, ZU_A, GU_SUM, EU_SUM, (size_t)NU_);
  apply_accum_kernel<<<2048, 256, 0, stream>>>(v_mul_s, T_U, ZI_A, GI_SUM, EI_SUM, (size_t)NI_);

  // ---- layer 2 (prev = ZU_A/ZI_A) -> ZU_B/ZI_B
  fill_zero_kernel<<<2048, 256, 0, stream>>>(ZU_B, EU + EI);
  fill_zero_kernel<<<4, 256, 0, stream>>>(T_I, 1024);
  spmm_kernel<<<2048, 256, 0, stream>>>(ZI_A, ZU_B, adj_v, adj_r, adj_c, NEDGE, 4u);
  spmm_kernel<<<2048, 256, 0, stream>>>(ZU_A, ZI_B, adj_v, adj_c, adj_r, NEDGE, 5u);
  lowrank_reduce_kernel<<<512, 256, 0, stream>>>(vt, ZI_A, T_I, NI_);
  lowrank_reduce_kernel<<<512, 256, 0, stream>>>(ut, ZU_A, T_U, NU_);
  apply_accum_kernel<<<2048, 256, 0, stream>>>(u_mul_s, T_I, ZU_B, GU_SUM, EU_SUM, (size_t)NU_);
  apply_accum_kernel<<<2048, 256, 0, stream>>>(v_mul_s, T_U, ZI_B, GI_SUM, EI_SUM, (size_t)NI_);

  // ---- normalization (bf16 matrices for WMMA; f32 gathered rows for scalar terms)
  normalize_bf16_kernel<<<2048, 256, 0, stream>>>(EU_SUM, EU_NORM, (size_t)NU_);
  normalize_bf16_kernel<<<2048, 256, 0, stream>>>(EI_SUM, EI_NORM, (size_t)NI_);
  gather_norm_kernel<<<128, 256, 0, stream>>>(GU_SUM, EU_SUM, uids, GSELB_U, GSELF_U, ESELF_U, BB);
  gather_norm_kernel<<<128, 256, 0, stream>>>(GI_SUM, EI_SUM, iids, GSELB_I, GSELF_I, ESELF_I, BB);

  // ---- zero rowsums + scalar accumulators (contiguous 1024+1024+16 floats)
  fill_zero_kernel<<<9, 256, 0, stream>>>(RS_U, (size_t)(BB + BB + 16));

  // ---- compute-bound logsumexp GEMMs via V_WMMA_F32_16X16X32_BF16 (double-buffered)
  negscore_wmma_kernel<<<dim3(16, BB / 16), 256, 0, stream>>>(GSELB_U, EU_NORM, RS_U, NU_ / 16);
  negscore_wmma_kernel<<<dim3(8,  BB / 16), 256, 0, stream>>>(GSELB_I, EI_NORM, RS_I, NI_ / 16);

  // ---- scalar loss pieces
  pos_score_kernel<<<128, 256, 0, stream>>>(GSELF_U, ESELF_U, &SCAL[0]);
  pos_score_kernel<<<128, 256, 0, stream>>>(GSELF_I, ESELF_I, &SCAL[1]);
  bpr_kernel<<<128, 256, 0, stream>>>(EU_SUM, EI_SUM, uids, pos, neg, &SCAL[2]);
  sumsq_kernel<<<1024, 256, 0, stream>>>(E_u_0, EU, &SCAL[3]);
  sumsq_kernel<<<1024, 256, 0, stream>>>(E_i_0, EI, &SCAL[3]);

  finalize_kernel<<<1, 256, 0, stream>>>(RS_U, RS_I, SCAL, out);
}
